// GCN_EG_59536836657518
// MI455X (gfx1250) — matrix operand order repaired and checked by reading, not compiled, not run on hardware
//
#include <hip/hip_runtime.h>

typedef __attribute__((ext_vector_type(8)))  _Float16 v8h;
typedef __attribute__((ext_vector_type(16))) _Float16 v16h;
typedef __attribute__((ext_vector_type(8)))  float    v8f;

// ---------------------------------------------------------------------------
// WMMA helpers (CDNA5 wave32, 16x16x32 f16 -> f32 accumulate)
// ---------------------------------------------------------------------------
__device__ __forceinline__ v8f wmma_f16(v16h a, v16h b, v8f c) {
  // 8 args: (neg_a, A, neg_b, B, c_mod, C, reuse_a, reuse_b)
  return __builtin_amdgcn_wmma_f32_16x16x32_f16(false, a, false, b, (short)0, c,
                                                false, false);
}

// Load one 16x32 f16 A-fragment from a row-major LDS tile [16][rowStride].
// ISA A layout (16-bit, 16x32): lanes 0-15 hold row M=lane, K = {0..7, 16..23};
// lanes 16-31 hold row M=lane-16, K = {8..15, 24..31}.
__device__ __forceinline__ v16h load_a_frag(const _Float16* tile, int rowStride,
                                            int ks, int lane) {
  int row = lane & 15;
  int hi  = lane >> 4;
  const _Float16* p = tile + row * rowStride + ks * 32 + hi * 8;
  v8h a0 = *(const v8h*)(p);
  v8h a1 = *(const v8h*)(p + 16);
  v16h r;
#pragma unroll
  for (int t = 0; t < 8; ++t) { r[t] = a0[t]; r[t + 8] = a1[t]; }
  return r;
}

__device__ __forceinline__ float silu(float x) {
  return x / (1.f + __expf(-x));
}

// ---------------------------------------------------------------------------
// Weight prep: fp32 row-major W[K][Nout] -> f16 WMMA-B tiles, zero padded.
// B layout per 32x16 tile: element t = tile*512 + lane*16 + tt holds
// W[ks*32 + (lane>=16?8:0) + (tt&7) + (tt>=8?16:0)][nt*16 + lane%16].
// ---------------------------------------------------------------------------
__global__ void prep_weights(const float* __restrict__ W, _Float16* __restrict__ out,
                             int K, int Nout, int KP, int NP) {
  int total = (KP >> 5) * (NP >> 4) * 512;
  for (int t = blockIdx.x * blockDim.x + threadIdx.x; t < total;
       t += gridDim.x * blockDim.x) {
    int tile = t >> 9;
    int rem  = t & 511;
    int lane = rem >> 4;
    int tt   = rem & 15;
    int ntiles = NP >> 4;
    int ks = tile / ntiles;
    int nt = tile - ks * ntiles;
    int col = (nt << 4) + (lane & 15);
    int k   = (ks << 5) + ((lane >> 4) << 3) + (tt & 7) + ((tt >> 3) << 4);
    float v = (k < K && col < Nout) ? W[k * Nout + col] : 0.f;
    out[t] = (_Float16)v;
  }
}

__global__ void zero_f32(float* __restrict__ p, int n) {
  int t = blockIdx.x * blockDim.x + threadIdx.x;
  if (t < n) p[t] = 0.f;
}

// ---------------------------------------------------------------------------
// Edge MLP: per wave, 16 edges.  e_in = [feats[i], feats[j], ea, |rel|^2]
// m_ij = silu(silu(e_in @ W1 + b1) @ W2 + b2);  m_i[i] += m_ij (atomic).
// K1P = padded ein (mult of 32), H = 2*ein actual, HP = padded hidden
// (mult of 32), output M = 32.
// ---------------------------------------------------------------------------
__global__ void __launch_bounds__(128) egnn_edge(
    const int* __restrict__ eidx, const float* __restrict__ ea,
    const float* __restrict__ pos, const float* __restrict__ featsIn, int fi,
    int K1P, int H, int HP, const _Float16* __restrict__ W1,
    const float* __restrict__ b1, const _Float16* __restrict__ W2,
    const float* __restrict__ b2, float* __restrict__ mI, int E) {
  __shared__ __align__(16) _Float16 s_ein[4][16 * 96];
  __shared__ __align__(16) _Float16 s_hid[4][16 * 160];
  __shared__ int   s_i[4][16];
  __shared__ int   s_j[4][16];
  __shared__ float s_sc[4][32];

  int w = threadIdx.x >> 5;
  int lane = threadIdx.x & 31;
  int eBase = (blockIdx.x * 4 + w) * 16;
  _Float16* pe = s_ein[w];
  _Float16* ph = s_hid[w];

  if (lane < 16) {
    int e = eBase + lane;
    if (e >= E) e = E - 1;
    int j = eidx[e];
    int i = eidx[E + e];
    s_i[w][lane] = i;
    s_j[w][lane] = j;
    float dx = pos[2 * j] - pos[2 * i];
    float dy = pos[2 * j + 1] - pos[2 * i + 1];
    s_sc[w][2 * lane] = ea[e];
    s_sc[w][2 * lane + 1] = dx * dx + dy * dy;
  }
  // Build f16 e_in tile [16][K1P], zero padded (same-wave LDS is in-order).
  for (int t = lane; t < 16 * K1P; t += 32) {
    int m = t / K1P;
    int k = t - m * K1P;
    int i = s_i[w][m];
    int j = s_j[w][m];
    float v = 0.f;
    if (k < fi)            v = featsIn[i * fi + k];
    else if (k < 2 * fi)   v = featsIn[j * fi + (k - fi)];
    else if (k == 2 * fi)  v = s_sc[w][2 * m];
    else if (k == 2 * fi + 1) v = s_sc[w][2 * m + 1];
    pe[t] = (_Float16)v;
  }

  // MLP1: [16 x K1P] @ [K1P x HP]  + bias, SiLU -> hidden LDS tile
  int nt1 = HP >> 4, ks1 = K1P >> 5;
  for (int nt = 0; nt < nt1; ++nt) {
    v8f acc = {};
    for (int ks = 0; ks < ks1; ++ks) {
      v16h a = load_a_frag(pe, K1P, ks, lane);
      v16h b = *(const v16h*)(W1 + ((ks * nt1 + nt) << 9) + (lane << 4));
      acc = wmma_f16(a, b, acc);
    }
    int n = (nt << 4) + (lane & 15);
    float bias = (n < H) ? b1[n] : 0.f;
    int mb = (lane >> 4) << 3;
#pragma unroll
    for (int r = 0; r < 8; ++r) {
      float xv = acc[r] + bias;
      ph[(mb + r) * HP + n] = (_Float16)((n < H) ? silu(xv) : 0.f);
    }
  }

  // MLP2: [16 x HP] @ [HP x 32] + bias, SiLU -> atomic segment-sum scatter
  int ks2 = HP >> 5;
  for (int nt = 0; nt < 2; ++nt) {
    v8f acc = {};
    for (int ks = 0; ks < ks2; ++ks) {
      v16h a = load_a_frag(ph, HP, ks, lane);
      v16h b = *(const v16h*)(W2 + ((ks * 2 + nt) << 9) + (lane << 4));
      acc = wmma_f16(a, b, acc);
    }
    int n = (nt << 4) + (lane & 15);
    float bias = b2[n];
    int mb = (lane >> 4) << 3;
#pragma unroll
    for (int r = 0; r < 8; ++r) {
      int e = eBase + mb + r;
      if (e < E) {
        float sv = silu(acc[r] + bias);
        atomicAdd(&mI[s_i[w][mb + r] * 32 + n], sv);
      }
    }
  }
}

// ---------------------------------------------------------------------------
// Node MLP: per wave, 16 nodes.  in = [feats (fi), m_i (32)] padded to K=64.
// h = silu(in @ W1 + b1) @ W2 + b2  (hidden = 64, out = 32, no final act).
// ---------------------------------------------------------------------------
__global__ void __launch_bounds__(128) egnn_node(
    const float* __restrict__ featsIn, int fi, const float* __restrict__ mI,
    const _Float16* __restrict__ W1, const float* __restrict__ b1,
    const _Float16* __restrict__ W2, const float* __restrict__ b2,
    float* __restrict__ featsOut, int N) {
  __shared__ __align__(16) _Float16 s_in[4][16 * 64];
  __shared__ __align__(16) _Float16 s_hid[4][16 * 64];
  int w = threadIdx.x >> 5;
  int lane = threadIdx.x & 31;
  int nBase = (blockIdx.x * 4 + w) * 16;
  _Float16* pi = s_in[w];
  _Float16* ph = s_hid[w];

  int KIN = fi + 32;
  for (int t = lane; t < 16 * 64; t += 32) {
    int m = t >> 6, k = t & 63;
    int node = nBase + m;
    float v = 0.f;
    if (node < N) {
      if (k < fi)       v = featsIn[node * fi + k];
      else if (k < KIN) v = mI[node * 32 + (k - fi)];
    }
    pi[t] = (_Float16)v;
  }

#pragma unroll
  for (int nt = 0; nt < 4; ++nt) {
    v8f acc = {};
#pragma unroll
    for (int ks = 0; ks < 2; ++ks) {
      v16h a = load_a_frag(pi, 64, ks, lane);
      v16h b = *(const v16h*)(W1 + ((ks * 4 + nt) << 9) + (lane << 4));
      acc = wmma_f16(a, b, acc);
    }
    int n = (nt << 4) + (lane & 15);
    float bias = b1[n];
    int mb = (lane >> 4) << 3;
#pragma unroll
    for (int r = 0; r < 8; ++r)
      ph[(mb + r) * 64 + n] = (_Float16)silu(acc[r] + bias);
  }

#pragma unroll
  for (int nt = 0; nt < 2; ++nt) {
    v8f acc = {};
#pragma unroll
    for (int ks = 0; ks < 2; ++ks) {
      v16h a = load_a_frag(ph, 64, ks, lane);
      v16h b = *(const v16h*)(W2 + ((ks * 2 + nt) << 9) + (lane << 4));
      acc = wmma_f16(a, b, acc);
    }
    int n = (nt << 4) + (lane & 15);
    float bias = b2[n];
    int mb = (lane >> 4) << 3;
#pragma unroll
    for (int r = 0; r < 8; ++r) {
      int node = nBase + mb + r;
      if (node < N) featsOut[node * 32 + n] = acc[r] + bias;
    }
  }
}

// ---------------------------------------------------------------------------
// Pooling: segment mean over batch, then @ linW + linb
// ---------------------------------------------------------------------------
__global__ void pool_accum(const float* __restrict__ feats,
                           const int* __restrict__ batch,
                           float* __restrict__ pooled, float* __restrict__ cnt,
                           int N) {
  int t = blockIdx.x * blockDim.x + threadIdx.x;
  if (t < N * 32) {
    int n = t >> 5, c = t & 31;
    int g = batch[n];
    atomicAdd(&pooled[g * 32 + c], feats[t]);
    if (c == 0) atomicAdd(&cnt[g], 1.f);
  }
}

__global__ void pool_final(const float* __restrict__ pooled,
                           const float* __restrict__ cnt,
                           const float* __restrict__ linW,
                           const float* __restrict__ linb,
                           float* __restrict__ out, int G) {
  int g = blockIdx.x * blockDim.x + threadIdx.x;
  if (g < G) {
    float c = fmaxf(cnt[g], 1.f);
    float s = 0.f;
#pragma unroll
    for (int i = 0; i < 32; ++i) s += (pooled[g * 32 + i] / c) * linW[i];
    out[g] = s + linb[0];
  }
}

// ---------------------------------------------------------------------------
// Host-side orchestration
// ---------------------------------------------------------------------------
extern "C" void kernel_launch(void* const* d_in, const int* in_sizes, int n_in,
                              void* d_out, int out_size, void* d_ws,
                              size_t ws_size, hipStream_t stream) {
  const float* x    = (const float*)d_in[0];
  const float* pos  = (const float*)d_in[1];
  const int*   eidx = (const int*)d_in[2];
  const float* ea   = (const float*)d_in[3];
  const int*   batch = (const int*)d_in[4];
  const float* linW = (const float*)d_in[29];
  const float* linb = (const float*)d_in[30];
  float* out = (float*)d_out;

  const int N = in_sizes[0] / 2;
  const int E = in_sizes[3];
  const int G = out_size;

  char* ws = (char*)d_ws;
  size_t off = 0;
  auto alloc = [&](size_t bytes) -> void* {
    void* p = ws + off;
    off = (off + bytes + 255) & ~(size_t)255;
    return p;
  };
  float* featsA = (float*)alloc((size_t)N * 32 * 4);
  float* featsB = (float*)alloc((size_t)N * 32 * 4);
  float* mI     = (float*)alloc((size_t)N * 32 * 4);
  float* pooled = (float*)alloc((size_t)G * 32 * 4);
  float* cnt    = (float*)alloc((size_t)G * 4);

  // Padded dims per layer
  const int fiL[3]  = {2, 32, 32};
  const int einL[3] = {6, 66, 66};     // 2*fi + 2
  const int K1P[3]  = {32, 96, 96};    // padded ein (mult of 32)
  const int HL[3]   = {12, 132, 132};  // 2*ein
  const int HP[3]   = {32, 160, 160};  // padded hidden (mult of 32)

  _Float16* eW1h[3]; _Float16* eW2h[3]; _Float16* nW1h[3]; _Float16* nW2h[3];
  auto tilesBytes = [](int KP, int NP) {
    return (size_t)(KP / 32) * (NP / 16) * 512 * sizeof(_Float16);
  };
  for (int l = 0; l < 3; ++l) {
    eW1h[l] = (_Float16*)alloc(tilesBytes(K1P[l], HP[l]));
    eW2h[l] = (_Float16*)alloc(tilesBytes(HP[l], 32));
    nW1h[l] = (_Float16*)alloc(tilesBytes(64, 64));
    nW2h[l] = (_Float16*)alloc(tilesBytes(64, 32));
  }
  (void)ws_size;

  auto prep = [&](const float* W, _Float16* o, int K, int Nout, int KP, int NP) {
    int total = (KP / 32) * (NP / 16) * 512;
    prep_weights<<<(total + 255) / 256, 256, 0, stream>>>(W, o, K, Nout, KP, NP);
  };
  for (int l = 0; l < 3; ++l) {
    int base = 5 + 8 * l;
    prep((const float*)d_in[base + 0], eW1h[l], einL[l], HL[l], K1P[l], HP[l]);
    prep((const float*)d_in[base + 2], eW2h[l], HL[l], 32, HP[l], 32);
    prep((const float*)d_in[base + 4], nW1h[l], fiL[l] + 32, 64, 64, 64);
    prep((const float*)d_in[base + 6], nW2h[l], 64, 32, 64, 64);
  }

  const float* featsCur = x;
  int fiCur = 2;
  float* nodeOut[3] = {featsA, featsB, featsA};

  for (int l = 0; l < 3; ++l) {
    int base = 5 + 8 * l;
    const float* eb1 = (const float*)d_in[base + 1];
    const float* eb2 = (const float*)d_in[base + 3];
    const float* nb1 = (const float*)d_in[base + 5];
    const float* nb2 = (const float*)d_in[base + 7];

    zero_f32<<<(N * 32 + 255) / 256, 256, 0, stream>>>(mI, N * 32);

    egnn_edge<<<(E + 63) / 64, 128, 0, stream>>>(
        eidx, ea, pos, featsCur, fiCur, K1P[l], HL[l], HP[l], eW1h[l], eb1,
        eW2h[l], eb2, mI, E);

    egnn_node<<<(N + 63) / 64, 128, 0, stream>>>(
        featsCur, fiCur, mI, nW1h[l], nb1, nW2h[l], nb2, nodeOut[l], N);

    featsCur = nodeOut[l];
    fiCur = 32;
  }

  zero_f32<<<(G * 32 + 255) / 256, 256, 0, stream>>>(pooled, G * 32);
  zero_f32<<<(G + 255) / 256, 256, 0, stream>>>(cnt, G);
  pool_accum<<<(N * 32 + 255) / 256, 256, 0, stream>>>(featsCur, batch, pooled,
                                                       cnt, N);
  pool_final<<<(G + 63) / 64, 64, 0, stream>>>(pooled, cnt, linW, linb, out, G);
}
